// AttentionBlock_60644938220068
// MI455X (gfx1250) — matrix operand order repaired and checked by reading, not compiled, not run on hardware
//
#include <hip/hip_runtime.h>
#include <math.h>

#define BB   2
#define LL   2048
#define DMM  512
#define HH   8
#define DD   64
#define NKK  40
#define NQQ  40
#define EPSF 1e-5f
#define NEGBIG -3.402823466e38f

typedef __attribute__((ext_vector_type(16))) _Float16 v16h;
typedef __attribute__((ext_vector_type(8)))  _Float16 v8h;
typedef __attribute__((ext_vector_type(8)))  float    v8f;

__device__ __forceinline__ v8f wmma16(v16h a, v16h b, v8f c) {
  // D = A(16x32 f16) x B(32x16 f16) + C(16x16 f32)
  return __builtin_amdgcn_wmma_f32_16x16x32_f16(
      /*neg_a=*/false, a, /*neg_b=*/false, b,
      /*c_mod=*/(short)0, c, /*reuse_a=*/false, /*reuse_b=*/false);
}

// A-fragment layout (16-bit A 16x32): element e of lane (h = lane>>4): k = 16h + e
__device__ __forceinline__ v16h loadA_f32src(const float* __restrict__ p, int h) {
  const float* q = p + 16 * h;
  float4 f0 = *(const float4*)(q);
  float4 f1 = *(const float4*)(q + 4);
  float4 f2 = *(const float4*)(q + 8);
  float4 f3 = *(const float4*)(q + 12);
  v16h a;
  a[0]=(_Float16)f0.x;  a[1]=(_Float16)f0.y;  a[2]=(_Float16)f0.z;  a[3]=(_Float16)f0.w;
  a[4]=(_Float16)f1.x;  a[5]=(_Float16)f1.y;  a[6]=(_Float16)f1.z;  a[7]=(_Float16)f1.w;
  a[8]=(_Float16)f2.x;  a[9]=(_Float16)f2.y;  a[10]=(_Float16)f2.z; a[11]=(_Float16)f2.w;
  a[12]=(_Float16)f3.x; a[13]=(_Float16)f3.y; a[14]=(_Float16)f3.z; a[15]=(_Float16)f3.w;
  return a;
}

// B-fragment layout (16-bit B 32x16): element e of lane (h = lane>>4): k = 16h + e
// f32 source, per-lane contiguous 16 floats
__device__ __forceinline__ v16h loadB_f32src(const float* __restrict__ p, int h) {
  const float* q = p + 16 * h;
  float4 f0 = *(const float4*)(q);
  float4 f1 = *(const float4*)(q + 4);
  float4 f2 = *(const float4*)(q + 8);
  float4 f3 = *(const float4*)(q + 12);
  v16h b;
  b[0]=(_Float16)f0.x;  b[1]=(_Float16)f0.y;  b[2]=(_Float16)f0.z;  b[3]=(_Float16)f0.w;
  b[4]=(_Float16)f1.x;  b[5]=(_Float16)f1.y;  b[6]=(_Float16)f1.z;  b[7]=(_Float16)f1.w;
  b[8]=(_Float16)f2.x;  b[9]=(_Float16)f2.y;  b[10]=(_Float16)f2.z; b[11]=(_Float16)f2.w;
  b[12]=(_Float16)f3.x; b[13]=(_Float16)f3.y; b[14]=(_Float16)f3.z; b[15]=(_Float16)f3.w;
  return b;
}

// B-fragment from pre-converted f16 memory: 16 contiguous halves at p+16h
// -> exactly two global_load_b128
__device__ __forceinline__ v16h loadB_f16src(const _Float16* __restrict__ p, int h) {
  v8h lo = *(const v8h*)(p + 16 * h);
  v8h hi = *(const v8h*)(p + 16 * h + 8);
  v16h b;
#pragma unroll
  for (int e = 0; e < 8; ++e) { b[e] = lo[e]; b[e + 8] = hi[e]; }
  return b;
}

// ---------------------------------------------------------------------------
// K0: Wt[z][c][k] = (f16) W[z][k][c]  (transpose+convert; W is L2-resident)
// ---------------------------------------------------------------------------
__global__ __launch_bounds__(256) void convw_kernel(
    const float* __restrict__ Wq, const float* __restrict__ Wk,
    const float* __restrict__ Wv, _Float16* __restrict__ Wt)
{
  int z = blockIdx.y;
  const float* W = (z == 0) ? Wq : ((z == 1) ? Wk : Wv);
  int idx = blockIdx.x * 256 + threadIdx.x;   // c*512 + k
  int c = idx >> 9, kx = idx & (DMM - 1);
  Wt[(size_t)z * DMM * DMM + idx] = (_Float16)W[(size_t)kx * DMM + c];
}

// ---------------------------------------------------------------------------
// K1: P = X @ W + bias, written in [B][H][L][D] layout. blockIdx.z selects q/k/v.
// Block 256 = 8 waves: 2(M) x 4(N); wave tile 16(M) x 32(N); K loop 512/32.
// B fragments come from Wt (f16, transposed): 2 x b128 per fragment.
// ---------------------------------------------------------------------------
__global__ __launch_bounds__(256) void proj_kernel(
    const float* __restrict__ q, const float* __restrict__ k, const float* __restrict__ v,
    const float* __restrict__ bq, const float* __restrict__ bk, const float* __restrict__ bv,
    const _Float16* __restrict__ Wt,
    float* __restrict__ qp, float* __restrict__ kp, float* __restrict__ vp)
{
  const float* X; const float* bias; float* P;
  if (blockIdx.z == 0)      { X = q; bias = bq; P = qp; }
  else if (blockIdx.z == 1) { X = k; bias = bk; P = kp; }
  else                      { X = v; bias = bv; P = vp; }
  const _Float16* Wth = Wt + (size_t)blockIdx.z * DMM * DMM;

  const int lane = threadIdx.x & 31;
  const int wave = threadIdx.x >> 5;
  const int wm = wave >> 2;                 // 0..1
  const int wn = wave & 3;                  // 0..3
  const int rowBase = blockIdx.x * 32 + wm * 16;
  const int colBase = blockIdx.y * 128 + wn * 32;
  const int ln = lane & 15;
  const int h  = lane >> 4;

  v8f c0 = {}; v8f c1 = {};
  const float*    Xr  = X   + (size_t)(rowBase + ln) * DMM;
  const _Float16* Wr0 = Wth + (size_t)(colBase + ln) * DMM;        // B col for c0
  const _Float16* Wr1 = Wth + (size_t)(colBase + 16 + ln) * DMM;   // B col for c1

  for (int kk = 0; kk < DMM; kk += 32) {
    if (kk + 32 < DMM) __builtin_prefetch(Xr + kk + 32, 0, 1);  // global_prefetch_b8
    v16h a  = loadA_f32src(Xr + kk, h);
    v16h b0 = loadB_f16src(Wr0 + kk, h);
    v16h b1 = loadB_f16src(Wr1 + kk, h);
    c0 = wmma16(a, b0, c0);
    c1 = wmma16(a, b1, c1);
  }

#pragma unroll
  for (int r = 0; r < 8; ++r) {
    int gr = rowBase + 8 * h + r;           // global row in [0, B*L)
    int bidx = gr >> 11;                    // / L
    int i    = gr & (LL - 1);
    {
      int gc = colBase + ln;
      int hh = gc >> 6, d = gc & 63;
      P[((size_t)(bidx * HH + hh) * LL + i) * DD + d] = c0[r] + bias[gc];
    }
    {
      int gc = colBase + 16 + ln;
      int hh = gc >> 6, d = gc & 63;
      P[((size_t)(bidx * HH + hh) * LL + i) * DD + d] = c1[r] + bias[gc];
    }
  }
}

// ---------------------------------------------------------------------------
// K2: measure[b,h,i] = max_j pre[j] - sum_j pre[j] / L, pre over 40 sampled keys
// ---------------------------------------------------------------------------
__global__ __launch_bounds__(256) void score_kernel(
    const float* __restrict__ qp, const float* __restrict__ kp,
    const int* __restrict__ ridx, float* __restrict__ measure)
{
  int idx = blockIdx.x * 256 + threadIdx.x;     // 0 .. B*H*L-1
  int bh = idx / LL;
  int i  = idx & (LL - 1);
  const float4* qr4 = (const float4*)(qp + (size_t)(bh * LL + i) * DD);
  float4 qr[16];
#pragma unroll
  for (int t = 0; t < 16; ++t) qr[t] = qr4[t];

  float mx = NEGBIG, sm = 0.f;
  for (int j = 0; j < NKK; ++j) {
    int jr = ridx[i * NKK + j];
    const float4* kr = (const float4*)(kp + (size_t)(bh * LL + jr) * DD);
    float dot = 0.f;
#pragma unroll
    for (int t = 0; t < 16; ++t) {
      float4 kf = kr[t];
      dot += qr[t].x * kf.x + qr[t].y * kf.y + qr[t].z * kf.z + qr[t].w * kf.w;
    }
    mx = fmaxf(mx, dot);
    sm += dot;
  }
  measure[idx] = mx - sm * (1.0f / (float)LL);
}

// ---------------------------------------------------------------------------
// K3: per (b,h): top-40 of measure (desc, ties -> lower index, jax semantics),
// plus inverse scatter map selpos (-1 elsewhere).
// ---------------------------------------------------------------------------
__global__ __launch_bounds__(256) void topk_kernel(
    const float* __restrict__ measure, int* __restrict__ qidx, int* __restrict__ selpos)
{
  __shared__ float sv[LL];
  __shared__ float rv[256];
  __shared__ int   ri[256];
  __shared__ int   sq[NQQ];
  int bh = blockIdx.x;
  int t  = threadIdx.x;
  for (int i = t; i < LL; i += 256) {
    sv[i] = measure[bh * LL + i];
    selpos[bh * LL + i] = -1;
  }
  __syncthreads();
  for (int n = 0; n < NQQ; ++n) {
    float bv = NEGBIG; int bi = 0x7fffffff;
    for (int i = t; i < LL; i += 256) {
      float vv = sv[i];
      if (vv > bv || (vv == bv && i < bi)) { bv = vv; bi = i; }
    }
    rv[t] = bv; ri[t] = bi;
    __syncthreads();
    for (int s = 128; s > 0; s >>= 1) {
      if (t < s) {
        float ov = rv[t + s]; int oi = ri[t + s];
        if (ov > rv[t] || (ov == rv[t] && oi < ri[t])) { rv[t] = ov; ri[t] = oi; }
      }
      __syncthreads();
    }
    int best = ri[0];
    if (t == 0) { qidx[bh * NQQ + n] = best; sq[n] = best; }
    __syncthreads();
    if (t == 0) sv[best] = NEGBIG;
    __syncthreads();
  }
  if (t < NQQ) selpos[bh * LL + sq[t]] = t;
}

// ---------------------------------------------------------------------------
// K4: attn logits = q_sel @ kp^T * D^-0.5  ->  attn region of d_out.
// Per block (bh, j-chunk of 128): 8 waves, one 16-col tile each; M padded to 48.
// ---------------------------------------------------------------------------
__global__ __launch_bounds__(256) void attn_logits_kernel(
    const float* __restrict__ qp, const float* __restrict__ kp,
    const int* __restrict__ qidx, float* __restrict__ attn)
{
  int bh   = blockIdx.x;
  int wave = threadIdx.x >> 5;
  int lane = threadIdx.x & 31;
  int ln = lane & 15, h = lane >> 4;
  int j0 = (blockIdx.y * 8 + wave) * 16;

  // B fragments: column j0+ln of kp^T == row j0+ln of kp (contiguous in d)
  const float* kr = kp + (size_t)(bh * LL + j0 + ln) * DD;
  v16h b0 = loadB_f32src(kr, h);        // K = 0..31
  v16h b1 = loadB_f32src(kr + 32, h);   // K = 32..63

#pragma unroll
  for (int mt = 0; mt < 3; ++mt) {
    int sel  = mt * 16 + ln;
    int selc = (sel < NQQ) ? sel : 0;       // pad rows duplicate row 0 (unstored)
    int row  = qidx[bh * NQQ + selc];
    const float* qr = qp + (size_t)(bh * LL + row) * DD;
    v16h a0 = loadA_f32src(qr, h);
    v16h a1 = loadA_f32src(qr + 32, h);
    v8f c = {};
    c = wmma16(a0, b0, c);
    c = wmma16(a1, b1, c);
#pragma unroll
    for (int r = 0; r < 8; ++r) {
      int s = mt * 16 + 8 * h + r;
      if (s < NQQ)
        attn[((size_t)(bh * NQQ + s)) * LL + j0 + ln] = c[r] * 0.125f; // D^-0.5
    }
  }
}

// ---------------------------------------------------------------------------
// K5: in-place softmax over L for each of the B*H*NQ rows in d_out attn region
// ---------------------------------------------------------------------------
__global__ __launch_bounds__(256) void softmax_kernel(float* __restrict__ attn)
{
  __shared__ float red[256];
  float* p = attn + (size_t)blockIdx.x * LL;
  int t = threadIdx.x;
  float x[8];
  float mx = NEGBIG;
#pragma unroll
  for (int u = 0; u < 8; ++u) { x[u] = p[t + u * 256]; mx = fmaxf(mx, x[u]); }
  red[t] = mx; __syncthreads();
  for (int s = 128; s > 0; s >>= 1) { if (t < s) red[t] = fmaxf(red[t], red[t + s]); __syncthreads(); }
  mx = red[0]; __syncthreads();
  float sm = 0.f;
#pragma unroll
  for (int u = 0; u < 8; ++u) { x[u] = expf(x[u] - mx); sm += x[u]; }
  red[t] = sm; __syncthreads();
  for (int s = 128; s > 0; s >>= 1) { if (t < s) red[t] += red[t + s]; __syncthreads(); }
  float inv = 1.0f / red[0];
#pragma unroll
  for (int u = 0; u < 8; ++u) p[t + u * 256] = x[u] * inv;
}

// ---------------------------------------------------------------------------
// K6: out_sel = attn(40x2048) @ vp(2048x64).  12 waves: 3 M-tiles x 4 N-tiles.
// ---------------------------------------------------------------------------
__global__ __launch_bounds__(384) void outsel_kernel(
    const float* __restrict__ attn, const float* __restrict__ vp,
    float* __restrict__ outsel)
{
  int bh   = blockIdx.x;
  int wave = threadIdx.x >> 5;        // 0..11
  int lane = threadIdx.x & 31;
  int ln = lane & 15, h = lane >> 4;
  int mt = wave >> 2;                 // 0..2
  int nt = wave & 3;                  // 0..3
  int n0 = nt * 16;

  int sel  = mt * 16 + ln;
  int selc = (sel < NQQ) ? sel : 0;
  const float* arow  = attn + (size_t)(bh * NQQ + selc) * LL;
  const float* vbase = vp + (size_t)bh * LL * DD;

  v8f c = {};
  for (int kk = 0; kk < LL; kk += 32) {
    v16h a = loadA_f32src(arow + kk, h);
    v16h b;
#pragma unroll
    for (int e = 0; e < 16; ++e) {
      int krow = kk + 16 * h + e;
      b[e] = (_Float16)vbase[(size_t)krow * DD + n0 + ln];
    }
    c = wmma16(a, b, c);
  }
#pragma unroll
  for (int r = 0; r < 8; ++r) {
    int s = mt * 16 + 8 * h + r;
    if (s < NQQ) outsel[(size_t)(bh * NQQ + s) * DD + n0 + ln] = c[r];
  }
}

// ---------------------------------------------------------------------------
// K7: v_mean[b,h,d] = mean_i vp[b,h,i,d]
// ---------------------------------------------------------------------------
__global__ __launch_bounds__(64) void vmean_kernel(
    const float* __restrict__ vp, float* __restrict__ vmean)
{
  int bh = blockIdx.x, d = threadIdx.x;
  const float* p = vp + (size_t)bh * LL * DD + d;
  float s = 0.f;
  for (int i = 0; i < LL; ++i) s += p[(size_t)i * DD];
  vmean[bh * DD + d] = s * (1.0f / (float)LL);
}

// ---------------------------------------------------------------------------
// K8: context scatter + residual + LayerNorm -> yn (first d_out region)
// ---------------------------------------------------------------------------
__global__ __launch_bounds__(256) void final_kernel(
    const float* __restrict__ qin, const float* __restrict__ outsel,
    const float* __restrict__ vmean, const int* __restrict__ selpos,
    const float* __restrict__ gamma, const float* __restrict__ beta,
    float* __restrict__ yn)
{
  __shared__ float s1[256], s2[256];
  int row = blockIdx.x;               // b*L + i
  int b = row >> 11, i = row & (LL - 1);
  int t = threadIdx.x;
  float y[2];
#pragma unroll
  for (int u = 0; u < 2; ++u) {
    int cidx = t + u * 256;
    int h = cidx >> 6, d = cidx & 63;
    int bh = b * HH + h;
    int pos = selpos[bh * LL + i];
    float ctx = (pos >= 0) ? outsel[(size_t)(bh * NQQ + pos) * DD + d]
                           : vmean[bh * DD + d];
    y[u] = ctx + qin[(size_t)row * DMM + cidx];
  }
  s1[t] = y[0] + y[1];
  s2[t] = y[0] * y[0] + y[1] * y[1];
  __syncthreads();
  for (int s = 128; s > 0; s >>= 1) {
    if (t < s) { s1[t] += s1[t + s]; s2[t] += s2[t + s]; }
    __syncthreads();
  }
  float mu  = s1[0] * (1.0f / (float)DMM);
  float var = s2[0] * (1.0f / (float)DMM) - mu * mu;
  float inv = rsqrtf(var + EPSF);
#pragma unroll
  for (int u = 0; u < 2; ++u) {
    int cidx = t + u * 256;
    yn[(size_t)row * DMM + cidx] = (y[u] - mu) * inv * gamma[cidx] + beta[cidx];
  }
}

// ---------------------------------------------------------------------------
extern "C" void kernel_launch(void* const* d_in, const int* in_sizes, int n_in,
                              void* d_out, int out_size, void* d_ws, size_t ws_size,
                              hipStream_t stream)
{
  (void)in_sizes; (void)n_in; (void)out_size; (void)ws_size;
  const float* q     = (const float*)d_in[0];
  const float* k     = (const float*)d_in[1];
  const float* v     = (const float*)d_in[2];
  const float* Wq    = (const float*)d_in[3];
  const float* bq    = (const float*)d_in[4];
  const float* Wk    = (const float*)d_in[5];
  const float* bk    = (const float*)d_in[6];
  const float* Wv    = (const float*)d_in[7];
  const float* bv    = (const float*)d_in[8];
  const float* gamma = (const float*)d_in[9];
  const float* beta  = (const float*)d_in[10];
  const int*   ridx  = (const int*)d_in[11];

  float* yn   = (float*)d_out;
  float* attn = yn + (size_t)BB * LL * DMM;     // second tuple output

  float* ws      = (float*)d_ws;
  float* qp      = ws;
  float* kp      = qp + (size_t)BB * HH * LL * DD;
  float* vp      = kp + (size_t)BB * HH * LL * DD;
  float* measure = vp + (size_t)BB * HH * LL * DD;
  int*   qidx    = (int*)(measure + (size_t)BB * HH * LL);
  int*   selpos  = qidx + BB * HH * NQQ;
  float* outsel  = (float*)(selpos + (size_t)BB * HH * LL);
  float* vmean   = outsel + (size_t)BB * HH * NQQ * DD;
  _Float16* Wt   = (_Float16*)(vmean + BB * HH * DD);   // [3][512][512] f16

  convw_kernel<<<dim3((DMM * DMM) / 256, 3), 256, 0, stream>>>(Wq, Wk, Wv, Wt);
  proj_kernel<<<dim3((BB * LL) / 32, DMM / 128, 3), 256, 0, stream>>>(
      q, k, v, bq, bk, bv, Wt, qp, kp, vp);
  score_kernel<<<(BB * HH * LL) / 256, 256, 0, stream>>>(qp, kp, ridx, measure);
  topk_kernel<<<BB * HH, 256, 0, stream>>>(measure, qidx, selpos);
  attn_logits_kernel<<<dim3(BB * HH, LL / 128), 256, 0, stream>>>(qp, kp, qidx, attn);
  softmax_kernel<<<BB * HH * NQQ, 256, 0, stream>>>(attn);
  outsel_kernel<<<BB * HH, 384, 0, stream>>>(attn, vp, outsel);
  vmean_kernel<<<BB * HH, 64, 0, stream>>>(vp, vmean);
  final_kernel<<<BB * LL, 256, 0, stream>>>(q, outsel, vmean, selpos, gamma, beta, yn);
}
